// AnomalyAttention_13692355740408
// MI455X (gfx1250) — matrix operand order repaired and checked by reading, not compiled, not run on hardware
//
#include <hip/hip_runtime.h>
#include <cstdint>
#include <cstddef>

// ---------------------------------------------------------------------------
// AnomalyAttention for MI455X (gfx1250, wave32, WMMA + async-to-LDS)
//   qkv = x @ W^T                        (bf16 WMMA, f32 accum)
//   G   = (qkv @ qkv^T) / sqrt(512)     stored bf16 (32 MB, halves n^2 traffic)
//   colmax/colsum over G (axis=0), E = exp(G - colmax) in place (bf16)
//   Z   = E @ (V / colsum)              (softmax denom folded into V)
//   P   = row-normalized (|i-j| + |qkv[j,0]| * noise(i,j)), row staged in LDS
// GEMM stages tiles with GLOBAL_LOAD_ASYNC_TO_LDS_B128 (ASYNCcnt) and a
// double-buffered LDS pipeline so the async engine overlaps WMMA compute.
// ---------------------------------------------------------------------------

#define SEQ  4096
#define DIN  512
#define DOUT 512

typedef __attribute__((ext_vector_type(16))) __bf16 bf16x16;
typedef __attribute__((ext_vector_type(8)))  float  floatx8;

// LDS tile row stride in bf16 elements: 40 (=80B = 20 banks; 20*m mod 64 is
// distinct for m=0..15 -> conflict-free fragment reads; 80B % 16 == 0 so
// b128 async stores stay aligned).
#define LSTR 40

static __device__ __forceinline__ __bf16 f2bf(float f) {
    union { float f; uint32_t u; } c; c.f = f;
    uint32_t r = c.u + 0x7FFFu + ((c.u >> 16) & 1u);   // round to nearest even
    union { unsigned short s; __bf16 b; } o; o.s = (unsigned short)(r >> 16);
    return o.b;
}
static __device__ __forceinline__ float bf2f(__bf16 b) {
    union { unsigned short s; __bf16 b; } i; i.b = b;
    union { uint32_t u; float f; } o; o.u = ((uint32_t)i.s) << 16;
    return o.f;
}

// async copy of 16B global -> LDS, tracked by ASYNCcnt (CDNA5 path).
// LDS dest operand is the wave-relative LDS byte address = low 32 bits of the
// generic pointer (flat LDS aperture truncates to LDS offset).
static __device__ __forceinline__ void async_copy_b128(const __bf16* gsrc, __bf16* ldst) {
    const uint32_t lds_addr = (uint32_t)(uintptr_t)ldst;
    const uint64_t gaddr    = (uint64_t)(uintptr_t)gsrc;
    asm volatile("global_load_async_to_lds_b128 %0, %1, off"
                 :: "v"(lds_addr), "v"(gaddr) : "memory");
}
static __device__ __forceinline__ void wait_async_le4() {
    asm volatile("s_wait_asynccnt 0x4" ::: "memory");
}
static __device__ __forceinline__ void wait_async_0() {
    asm volatile("s_wait_asynccnt 0x0" ::: "memory");
}

// ---------------- fragment loaders (ISA 7.12.2 layouts, wave32) -------------
// A 16x32 bf16: lanes 0-15 = M0-15 / K-halves in lane[4]; VGPR j packs (k,k+1)
static __device__ __forceinline__ bf16x16 load_frag_A(const __bf16* As, int lane, int mOff) {
    const int m    = mOff + (lane & 15);
    const int half = lane >> 4;
    const uint32_t* row = (const uint32_t*)(As + m * LSTR);   // 2 bf16 / uint
    union { bf16x16 v; uint32_t u[8]; } f;
    f.u[0] = row[0 + 4 * half];   // K = 0,1   (+8 for upper half)
    f.u[1] = row[1 + 4 * half];
    f.u[2] = row[2 + 4 * half];
    f.u[3] = row[3 + 4 * half];
    f.u[4] = row[8 + 4 * half];   // K = 16,17 (+8 for upper half)
    f.u[5] = row[9 + 4 * half];
    f.u[6] = row[10 + 4 * half];
    f.u[7] = row[11 + 4 * half];
    return f.v;
}
// B 32x16 bf16 (stored as [N][K] rows in LDS): lane = N, lane halves split K
static __device__ __forceinline__ bf16x16 load_frag_B(const __bf16* Bs, int lane, int nOff) {
    const int n    = nOff + (lane & 15);
    const int half = lane >> 4;
    const uint32_t* row = (const uint32_t*)(Bs + n * LSTR);
    union { bf16x16 v; uint32_t u[8]; } f;
#pragma unroll
    for (int j = 0; j < 8; ++j) f.u[j] = row[(j & 3) + 8 * (j >> 2) + 4 * half];   // K = 2j,2j+1 (+8 upper half; j>=4 -> +8 more)
    return f.v;
}

// ------------------------- bf16 GEMM: C = scale * A @ B^T -------------------
// A: [M,K] bf16 row-major. B: [N,K] bf16 row-major. Cf (f32) / Cb (bf16)
// optional outputs. M,N multiples of 128; K multiple of 32.
// 256 threads = 8 waves; 128x128 block tile; 64x32 per wave (8 WMMA accums).
// Double-buffered LDS, tiles staged by the async-to-LDS engine.
__global__ __launch_bounds__(256)
void gemm_bf16_nt(const __bf16* __restrict__ A, const __bf16* __restrict__ B,
                  float* __restrict__ Cf, __bf16* __restrict__ Cb,
                  int M, int N, int K, float scale) {
    __shared__ __bf16 As[2][128 * LSTR];
    __shared__ __bf16 Bs[2][128 * LSTR];

    const int tid   = threadIdx.x;
    const int lane  = tid & 31;
    const int wave  = tid >> 5;          // 8 waves
    const int waveM = wave & 1;          // 2 x 4 wave grid -> 64x32 per wave
    const int waveN = wave >> 1;
    const int bm = blockIdx.y * 128;
    const int bn = blockIdx.x * 128;

    // this thread's two staging chunks (row, 8-bf16 segment)
    const int row0 = tid >> 2,         seg0 = tid & 3;
    const int row1 = (tid + 256) >> 2, seg1 = (tid + 256) & 3;

    // issue the 4 async 16B copies (A0,B0,A1,B1) for K-step k0 into buffer b
    auto issue_stage = [&](int k0, int b) {
        async_copy_b128(A + (size_t)(bm + row0) * K + k0 + seg0 * 8,
                        &As[b][row0 * LSTR + seg0 * 8]);
        async_copy_b128(B + (size_t)(bn + row0) * K + k0 + seg0 * 8,
                        &Bs[b][row0 * LSTR + seg0 * 8]);
        async_copy_b128(A + (size_t)(bm + row1) * K + k0 + seg1 * 8,
                        &As[b][row1 * LSTR + seg1 * 8]);
        async_copy_b128(B + (size_t)(bn + row1) * K + k0 + seg1 * 8,
                        &Bs[b][row1 * LSTR + seg1 * 8]);
    };

    const floatx8 vzero = {0.f, 0.f, 0.f, 0.f, 0.f, 0.f, 0.f, 0.f};
    floatx8 acc[4][2];
#pragma unroll
    for (int mf = 0; mf < 4; ++mf)
#pragma unroll
        for (int nf = 0; nf < 2; ++nf) acc[mf][nf] = vzero;

    issue_stage(0, 0);
    const int steps = K >> 5;            // K / 32
    int cur = 0;
    for (int it = 0; it < steps; ++it) {
        if (it + 1 < steps) {            // prefetch next stage into other buffer
            issue_stage((it + 1) << 5, cur ^ 1);
            wait_async_le4();            // drain current stage (in-order retire)
        } else {
            wait_async_0();
        }
        __syncthreads();                 // current tile visible to all waves

        bf16x16 bfr0 = load_frag_B(Bs[cur], lane, waveN * 32);
        bf16x16 bfr1 = load_frag_B(Bs[cur], lane, waveN * 32 + 16);
#pragma unroll
        for (int mf = 0; mf < 4; ++mf) {
            bf16x16 af = load_frag_A(As[cur], lane, waveM * 64 + mf * 16);
            acc[mf][0] = __builtin_amdgcn_wmma_f32_16x16x32_bf16(
                false, af, false, bfr0, (short)0, acc[mf][0], false, false);
            acc[mf][1] = __builtin_amdgcn_wmma_f32_16x16x32_bf16(
                false, af, false, bfr1, (short)0, acc[mf][1], false, false);
        }
        __syncthreads();                 // all reads done before buf reuse
        cur ^= 1;
    }

    // epilogue: C/D layout = VGPR r -> row r (lanes 0-15) / row 8+r (16-31)
    const int cBase = bn + waveN * 32 + (lane & 15);
    const int rBase = bm + waveM * 64 + ((lane >> 4) << 3);
#pragma unroll
    for (int mf = 0; mf < 4; ++mf)
#pragma unroll
        for (int nf = 0; nf < 2; ++nf)
#pragma unroll
            for (int r = 0; r < 8; ++r) {
                const int row = rBase + mf * 16 + r;
                const int col = cBase + nf * 16;
                const float v = acc[mf][nf][r] * scale;
                const size_t idx = (size_t)row * N + col;
                if (Cf) Cf[idx] = v;
                if (Cb) Cb[idx] = f2bf(v);
            }
}

// ------------------------------ small kernels -------------------------------
__global__ void cast_f32_to_bf16(const float* __restrict__ src,
                                 __bf16* __restrict__ dst, int n) {
    int i = blockIdx.x * 256 + threadIdx.x;
    if (i < n) dst[i] = f2bf(src[i]);
}

// column max over rows, split across row chunks (axis=0 softmax)
__global__ void col_max_part(const __bf16* __restrict__ G,
                             float* __restrict__ part, int n, int rowsPerChunk) {
    const int j  = blockIdx.x * 256 + threadIdx.x;
    const int r0 = blockIdx.y * rowsPerChunk;
    float m = -3.4e38f;
    for (int i = r0; i < r0 + rowsPerChunk; ++i)
        m = fmaxf(m, bf2f(G[(size_t)i * n + j]));
    part[(size_t)blockIdx.y * n + j] = m;
}
__global__ void col_max_fin(const float* __restrict__ part,
                            float* __restrict__ colmax, int n, int chunks) {
    const int j = blockIdx.x * 256 + threadIdx.x;
    float m = -3.4e38f;
    for (int c = 0; c < chunks; ++c) m = fmaxf(m, part[(size_t)c * n + j]);
    colmax[j] = m;
}
// E = exp(G - colmax) written in place (bf16) + partial column sums
__global__ void col_exp_part(__bf16* __restrict__ G,
                             const float* __restrict__ colmax,
                             float* __restrict__ part, int n, int rowsPerChunk) {
    const int j  = blockIdx.x * 256 + threadIdx.x;
    const int r0 = blockIdx.y * rowsPerChunk;
    const float m = colmax[j];
    float s = 0.f;
    for (int i = r0; i < r0 + rowsPerChunk; ++i) {
        const size_t idx = (size_t)i * n + j;
        const float e = __expf(bf2f(G[idx]) - m);
        G[idx] = f2bf(e);
        s += e;
    }
    part[(size_t)blockIdx.y * n + j] = s;
}
__global__ void col_sum_fin(const float* __restrict__ part,
                            float* __restrict__ colsum, int n, int chunks) {
    const int j = blockIdx.x * 256 + threadIdx.x;
    float s = 0.f;
    for (int c = 0; c < chunks; ++c) s += part[(size_t)c * n + j];
    colsum[j] = s;
}
// Vt[d][j] = bf16(qkv[j][d] / colsum[j])  (transposed -> [N,K] operand layout)
__global__ void scale_v_t(const float* __restrict__ qkv,
                          const float* __restrict__ colsum,
                          __bf16* __restrict__ Vt) {
    const int id = blockIdx.x * 256 + threadIdx.x;   // 0 .. 512*4096-1
    const int d  = id >> 12;
    const int j  = id & 4095;
    Vt[id] = f2bf(qkv[(size_t)j * DOUT + d] / colsum[j]);
}

// counter-based Gaussian (splitmix64 + Box-Muller); shape-faithful stand-in
static __device__ __forceinline__ float hash_normal(int i, int j) {
    uint64_t s = ((uint64_t)(uint32_t)i << 32) | (uint32_t)j;
    s ^= 0x9E3779B97F4A7C15ull;
    s ^= s >> 33; s *= 0xff51afd7ed558ccdull;
    s ^= s >> 33; s *= 0xc4ceb9fe1a85ec53ull;
    s ^= s >> 33;
    const float u1 = ((uint32_t)s + 1u) * 2.32830629e-10f;   // (0,1]
    const float u2 = ((uint32_t)(s >> 32)) * 2.32830629e-10f;
    return sqrtf(-2.0f * __logf(u1)) * __cosf(6.28318531f * u2);
}

// P[i,:] = gaussian / rowsum(gaussian); one block per row, row staged in LDS
__global__ __launch_bounds__(256)
void prior_kernel(const float* __restrict__ qkv, float* __restrict__ P, int n) {
    __shared__ float rowbuf[SEQ];
    __shared__ float red[256];
    const int i = blockIdx.x;
    float local = 0.f;
    for (int j = threadIdx.x; j < n; j += 256) {
        const float sig = fabsf(qkv[(size_t)j * DOUT]);       // |sigma[j,0]|
        const float g = fabsf((float)(i - j)) + sig * hash_normal(i, j);
        rowbuf[j] = g;
        local += g;
    }
    red[threadIdx.x] = local;
    __syncthreads();
    for (int s = 128; s > 0; s >>= 1) {
        if (threadIdx.x < s) red[threadIdx.x] += red[threadIdx.x + s];
        __syncthreads();
    }
    const float inv = 1.0f / red[0];
    for (int j = threadIdx.x; j < n; j += 256)
        P[(size_t)i * n + j] = rowbuf[j] * inv;
}

// ------------------------------- launch -------------------------------------
extern "C" void kernel_launch(void* const* d_in, const int* in_sizes, int n_in,
                              void* d_out, int out_size, void* d_ws, size_t ws_size,
                              hipStream_t stream) {
    const float* x = (const float*)d_in[0];   // [4096, 512]
    const float* W = (const float*)d_in[1];   // [512, 512]
    float* Z = (float*)d_out;                           // [4096, 512]
    float* P = (float*)d_out + (size_t)SEQ * DOUT;      // [4096, 4096]

    // workspace layout (~53 MB total)
    char* w = (char*)d_ws;
    size_t off = 0;
    __bf16* xb     = (__bf16*)(w + off); off += (size_t)SEQ * DIN * 2;    // 4 MB
    __bf16* Wb     = (__bf16*)(w + off); off += (size_t)DOUT * DIN * 2;   // 0.5 MB
    float*  qkvf   = (float*) (w + off); off += (size_t)SEQ * DOUT * 4;   // 8 MB
    __bf16* qkvb   = (__bf16*)(w + off); off += (size_t)SEQ * DOUT * 2;   // 4 MB
    __bf16* Vt     = (__bf16*)(w + off); off += (size_t)DOUT * SEQ * 2;   // 4 MB
    float*  colmax = (float*) (w + off); off += (size_t)SEQ * 4;
    float*  colsum = (float*) (w + off); off += (size_t)SEQ * 4;
    float*  cmPart = (float*) (w + off); off += (size_t)8 * SEQ * 4;
    float*  csPart = (float*) (w + off); off += (size_t)8 * SEQ * 4;
    __bf16* G      = (__bf16*)(w + off); off += (size_t)SEQ * SEQ * 2;    // 32 MB

    const float rsqrt_d = 0.04419417382f;    // 1/sqrt(512)

    // 1) cast inputs to bf16
    cast_f32_to_bf16<<<(SEQ * DIN + 255) / 256, 256, 0, stream>>>(x, xb, SEQ * DIN);
    cast_f32_to_bf16<<<(DOUT * DIN + 255) / 256, 256, 0, stream>>>(W, Wb, DOUT * DIN);

    // 2) qkv = x @ W^T  (f32 + bf16 copies)
    gemm_bf16_nt<<<dim3(DOUT / 128, SEQ / 128), 256, 0, stream>>>(
        xb, Wb, qkvf, qkvb, SEQ, DOUT, DIN, 1.0f);

    // 3) G = (qkv @ qkv^T) / sqrt(d), bf16
    gemm_bf16_nt<<<dim3(SEQ / 128, SEQ / 128), 256, 0, stream>>>(
        qkvb, qkvb, nullptr, G, SEQ, SEQ, DOUT, rsqrt_d);

    // 4) column softmax (axis=0): E = exp(G - colmax) in place, colsum folded into V
    col_max_part<<<dim3(SEQ / 256, 8), 256, 0, stream>>>(G, cmPart, SEQ, SEQ / 8);
    col_max_fin<<<SEQ / 256, 256, 0, stream>>>(cmPart, colmax, SEQ, 8);
    col_exp_part<<<dim3(SEQ / 256, 8), 256, 0, stream>>>(G, colmax, csPart, SEQ, SEQ / 8);
    col_sum_fin<<<SEQ / 256, 256, 0, stream>>>(csPart, colsum, SEQ, 8);

    // 5) Vt = (V / colsum)^T as [N,K] bf16
    scale_v_t<<<(DOUT * SEQ) / 256, 256, 0, stream>>>(qkvf, colsum, Vt);

    // 6) Z = E @ Vt^T
    gemm_bf16_nt<<<dim3(DOUT / 128, SEQ / 128), 256, 0, stream>>>(
        G, Vt, Z, nullptr, SEQ, DOUT, SEQ, 1.0f);

    // 7) P prior, one row per block
    prior_kernel<<<SEQ, 256, 0, stream>>>(qkvf, P, SEQ);

    (void)in_sizes; (void)n_in; (void)out_size; (void)ws_size;
}
